// Event_SMLP_Quantized_35381940585136
// MI455X (gfx1250) — matrix operand order repaired and checked by reading, not compiled, not run on hardware
//
#include <hip/hip_runtime.h>

typedef __attribute__((ext_vector_type(16))) _Float16 v16h;
typedef __attribute__((ext_vector_type(8)))  _Float16 v8h;
typedef __attribute__((ext_vector_type(4)))  _Float16 v4h;
typedef __attribute__((ext_vector_type(8)))  float    v8f;
typedef __attribute__((ext_vector_type(4)))  float    f32x4;
typedef __attribute__((ext_vector_type(4)))  unsigned int u32x4;
typedef __attribute__((ext_vector_type(8)))  int      i32x8;
typedef __attribute__((ext_vector_type(4)))  int      i32x4;

#define B_TOT  8192
#define K_IN   784
#define K_PAD  800           // 25 chunks of 32, zero padded
#define H_DIM  1024
#define O_DIM  10
#define O_PAD  16
#define THRESH 0.5f
#define DECAY  0.2f

#define BT     64            // batch rows per workgroup (128 WGs)
#define NCHUNK (K_PAD / 32)  // 25
#define XS1S   1024          // xs1 row stride (f16) -> 2048B rows, 16B aligned
#define WCH    (H_DIM * 32)  // one W1 K-chunk: [1024 rows][32 k] f16 = 64KB
#define XS1_ELEMS (BT * XS1S)
#define LDS_BYTES ((XS1_ELEMS + 3 * WCH) * 2)   // 128KB + 192KB = 320KB

// ---------------------------------------------------------------------------
// Weight conversion: W1 f32[1024x784] -> f16[1024x800] (K zero-padded),
//                    W2 f32[10x1024]  -> f16[16x1024]  (N zero-padded rows)
// ---------------------------------------------------------------------------
__global__ void convert_weights(const float* __restrict__ W1,
                                const float* __restrict__ W2,
                                _Float16* __restrict__ W1h,
                                _Float16* __restrict__ W2h) {
    int i = blockIdx.x * blockDim.x + threadIdx.x;
    if (i < H_DIM * K_PAD) {
        int r = i / K_PAD, c = i - r * K_PAD;
        W1h[i] = (c < K_IN) ? (_Float16)W1[r * K_IN + c] : (_Float16)0.f;
    }
    if (i < O_PAD * H_DIM) {
        int r = i / H_DIM, c = i - r * H_DIM;
        W2h[i] = (r < O_DIM) ? (_Float16)W2[r * H_DIM + c] : (_Float16)0.f;
    }
}

__device__ __forceinline__ v16h cat16(v8h lo, v8h hi) {
    return __builtin_shufflevector(lo, hi, 0,1,2,3,4,5,6,7,8,9,10,11,12,13,14,15);
}

#if __has_builtin(__builtin_amdgcn_tensor_load_to_lds)
#define HAVE_TDM 1
#else
#define HAVE_TDM 0
#endif

// Stage one [1024][32] f16 K-chunk of W1 into LDS buffer `buf` (0..2).
// TDM path: single asynchronous tensor DMA, tracked by TENSORcnt.
__device__ __forceinline__
void stage_w1_chunk(const _Float16* __restrict__ W1h, int chunk, int buf, int lane) {
    unsigned ldsOff = (unsigned)((XS1_ELEMS + buf * WCH) * 2);
#if HAVE_TDM
    unsigned long long ga = (unsigned long long)(size_t)(W1h + chunk * 32);
    u32x4 g0;
    g0[0] = 1u;                                                 // count=1, user mode
    g0[1] = ldsOff;                                             // lds_addr (bytes)
    g0[2] = (unsigned)ga;                                       // global_addr[31:0]
    g0[3] = (unsigned)((ga >> 32) & 0x01FFFFFFu) | (2u << 30);  // ga[56:32] | type=2
    i32x8 g1;
    g1[0] = 1 << 16;                          // workgroup_mask=0, data_size=2B
    g1[1] = (int)((unsigned)K_PAD << 16);     // tensor_dim0 = 800 (lo16)
    g1[2] = (int)((unsigned)H_DIM << 16);     // td0 hi=0 | tensor_dim1 = 1024 (lo16)
    g1[3] = (int)(32u << 16);                 // td1 hi=0 | tile_dim0 = 32
    g1[4] = H_DIM;                            // tile_dim1 = 1024
    g1[5] = K_PAD;                            // tensor_dim0_stride = 800
    g1[6] = 0;
    g1[7] = 0;
    i32x4 z4 = {};
    i32x8 z8 = {};
    // clang-23 / therock-10.0 signature: 6 args (g0, g1, g2, g3, extra, cpol)
    __builtin_amdgcn_tensor_load_to_lds(g0, g1, z4, z4, z8, 0);
#else
    // fallback: this wave copies the chunk with plain loads/stores
    extern __shared__ _Float16 smem[];
    _Float16* dst = smem + XS1_ELEMS + buf * WCH;
    for (int h = lane; h < H_DIM; h += 32) {
        const v8h* s = (const v8h*)(W1h + (size_t)h * K_PAD + chunk * 32);
        v8h* d = (v8h*)(dst + h * 32);
        d[0] = s[0]; d[1] = s[1]; d[2] = s[2]; d[3] = s[3];
    }
#endif
}

__device__ __forceinline__ void tdm_wait(int rem) {
#if HAVE_TDM
    if (rem >= 2)      __builtin_amdgcn_s_wait_tensorcnt(2);
    else if (rem == 1) __builtin_amdgcn_s_wait_tensorcnt(1);
    else               __builtin_amdgcn_s_wait_tensorcnt(0);
#endif
}

// ---------------------------------------------------------------------------
// Fused SNN: all T timesteps; membranes persistent in WMMA accumulators;
// W1 streamed through LDS by the Tensor Data Mover, triple-buffered.
// ---------------------------------------------------------------------------
__global__ void __launch_bounds__(256, 1)
snn_fused(const float* __restrict__ sample,     // [8192][784] f32
          const float* __restrict__ rand_u,     // [T][8192][784] f32
          const _Float16* __restrict__ W1h,     // [1024][800] f16
          const _Float16* __restrict__ W2h,     // [16][1024] f16
          float* __restrict__ out,              // [8192][10] f32
          int T) {
    extern __shared__ _Float16 smem[];
    _Float16* xs1 = smem;                       // [64][1024]: x (cols<800) then s1
    const _Float16* wb0 = smem + XS1_ELEMS;     // 3 x [1024][32] W1 chunk buffers

    const int tid    = threadIdx.x;
    const int lane   = tid & 31;
    const int wave   = tid >> 5;
    const int laneLo = lane & 15;
    const int hi     = (lane >> 4) & 1;
    const int bHalf  = wave & 1;     // which 32 batch rows
    const int hQuad  = wave >> 1;    // which 256-wide H slice
    const int wgB0   = blockIdx.x * BT;

    // Layer-1 membrane: 32 tiles of 16x16 f32 (two 16-row groups x 16 col tiles)
    v8f zero = {};
    v8f accA[16], accB[16];
#pragma unroll
    for (int i = 0; i < 16; ++i) { accA[i] = zero; accB[i] = zero; }

    // Layer-2 state (meaningful on waves with hQuad==0)
    v8f acc2A = zero, acc2B = zero;
    float sumA[8], sumB[8];
#pragma unroll
    for (int j = 0; j < 8; ++j) { sumA[j] = 0.f; sumB[j] = 0.f; }

    for (int t = 0; t < T; ++t) {
        __syncthreads();   // previous iteration's readers of xs1/wbuf are done

        if (wave == 0) {   // prime the TDM pipeline: chunks 0..2 in flight
            stage_w1_chunk(W1h, 0, 0, lane);
            stage_w1_chunk(W1h, 1, 1, lane);
            stage_w1_chunk(W1h, 2, 2, lane);
        }

        // ---- Bernoulli rate coding (float4-vectorized) -> f16 0/1 in LDS ----
        const float* rt = rand_u + (size_t)t * B_TOT * K_IN;
        for (int i = tid; i < BT * (K_IN / 4); i += 256) {
            int r  = i / (K_IN / 4);
            int c4 = i - r * (K_IN / 4);
            size_t g = (size_t)(wgB0 + r) * K_IN + c4 * 4;
            f32x4 sv = *(const f32x4*)(sample + g);
            f32x4 uv = *(const f32x4*)(rt + g);
            v4h x;
#pragma unroll
            for (int e = 0; e < 4; ++e)
                x[e] = sv[e] > uv[e] ? (_Float16)1.f : (_Float16)0.f;
            *(v4h*)(xs1 + r * XS1S + c4 * 4) = x;
        }
        {   // re-zero K pad cols 784..799 (s1 overwrote them last step)
            int r = tid >> 2, c4 = (K_IN / 4) + (tid & 3);
            v4h zp = {};
            *(v4h*)(xs1 + r * XS1S + c4 * 4) = zp;
        }
        __syncthreads();

        // ---- layer 1: m1 = decay*m1(reset) + x @ W1^T, TDM-pipelined ----
        const _Float16* xrA = xs1 + (bHalf * 32 +  0 + laneLo) * XS1S;
        const _Float16* xrB = xs1 + (bHalf * 32 + 16 + laneLo) * XS1S;
#pragma unroll 1
        for (int c = 0; c < NCHUNK; ++c) {
            if (wave == 0) tdm_wait(NCHUNK - 1 - c);  // chunk c landed in LDS
            __syncthreads();                          // broadcast readiness
            if (wave == 0 && c + 3 < NCHUNK)          // keep 3 chunks in flight
                stage_w1_chunk(W1h, c + 3, (c + 3) % 3, lane);

            const int k0 = c * 32;
            // A fragments: lanes0-15 K{0..7,16..23}, lanes16-31 K{8..15,24..31}
            v16h a0 = cat16(*(const v8h*)(xrA + k0 + (hi ? 8 : 0)),
                            *(const v8h*)(xrA + k0 + 16 + (hi ? 8 : 0)));
            v16h a1 = cat16(*(const v8h*)(xrB + k0 + (hi ? 8 : 0)),
                            *(const v8h*)(xrB + k0 + 16 + (hi ? 8 : 0)));
            const _Float16* bbase = wb0 + (c % 3) * WCH
                                  + (hQuad * 256 + laneLo) * 32 + (hi ? 8 : 0);
#pragma unroll
            for (int nt = 0; nt < 16; ++nt) {
                const _Float16* bp = bbase + nt * (16 * 32);
                v16h b = cat16(*(const v8h*)(bp), *(const v8h*)(bp + 16));
                accA[nt] = __builtin_amdgcn_wmma_f32_16x16x32_f16(
                    false, a0, false, b, (short)0, accA[nt], false, false);
                accB[nt] = __builtin_amdgcn_wmma_f32_16x16x32_f16(
                    false, a1, false, b, (short)0, accB[nt], false, false);
            }
        }
        __syncthreads();   // all GEMM reads of xs1 done; safe to write spikes

        // ---- spike, reset + pre-decay; s1 overwrites xs1 ----
#pragma unroll
        for (int nt = 0; nt < 16; ++nt) {
            int col = hQuad * 256 + nt * 16 + laneLo;
#pragma unroll
            for (int j = 0; j < 8; ++j) {
                int rowA = bHalf * 32 + (hi ? j + 8 : j);
                float mA = accA[nt][j];
                bool  sA = mA > THRESH;
                xs1[rowA * XS1S + col] = sA ? (_Float16)1.f : (_Float16)0.f;
                accA[nt][j] = sA ? 0.f : mA * DECAY;

                int rowB = rowA + 16;
                float mB = accB[nt][j];
                bool  sB = mB > THRESH;
                xs1[rowB * XS1S + col] = sB ? (_Float16)1.f : (_Float16)0.f;
                accB[nt][j] = sB ? 0.f : mB * DECAY;
            }
        }
        __syncthreads();

        // ---- layer 2 (waves 0,1; wave-uniform branch keeps EXEC full) ----
        if (hQuad == 0) {
            const _Float16* srA = xs1 + (bHalf * 32 +  0 + laneLo) * XS1S;
            const _Float16* srB = xs1 + (bHalf * 32 + 16 + laneLo) * XS1S;
            const _Float16* w2b = W2h + (size_t)laneLo * H_DIM + (hi ? 8 : 0);
            v8f c2A = acc2A, c2B = acc2B;
#pragma unroll 4
            for (int k0 = 0; k0 < H_DIM; k0 += 32) {
                v16h b  = cat16(*(const v8h*)(w2b + k0),
                                *(const v8h*)(w2b + k0 + 16));
                v16h a0 = cat16(*(const v8h*)(srA + k0 + (hi ? 8 : 0)),
                                *(const v8h*)(srA + k0 + 16 + (hi ? 8 : 0)));
                v16h a1 = cat16(*(const v8h*)(srB + k0 + (hi ? 8 : 0)),
                                *(const v8h*)(srB + k0 + 16 + (hi ? 8 : 0)));
                c2A = __builtin_amdgcn_wmma_f32_16x16x32_f16(
                    false, a0, false, b, (short)0, c2A, false, false);
                c2B = __builtin_amdgcn_wmma_f32_16x16x32_f16(
                    false, a1, false, b, (short)0, c2B, false, false);
            }
#pragma unroll
            for (int j = 0; j < 8; ++j) {
                float mA = c2A[j]; bool sA = mA > THRESH;
                sumA[j] += sA ? 1.f : 0.f;
                c2A[j] = sA ? 0.f : mA * DECAY;
                float mB = c2B[j]; bool sB = mB > THRESH;
                sumB[j] += sB ? 1.f : 0.f;
                c2B[j] = sB ? 0.f : mB * DECAY;
            }
            acc2A = c2A; acc2B = c2B;
        }
    }

    // ---- output: sumspike / T, [8192][10] f32 ----
    if (hQuad == 0 && laneLo < O_DIM) {
        float invT = 1.0f / (float)T;
#pragma unroll
        for (int j = 0; j < 8; ++j) {
            int rowA = wgB0 + bHalf * 32 + (hi ? j + 8 : j);
            out[(size_t)rowA * O_DIM + laneLo] = sumA[j] * invT;
            out[(size_t)(rowA + 16) * O_DIM + laneLo] = sumB[j] * invT;
        }
    }
}

// ---------------------------------------------------------------------------
extern "C" void kernel_launch(void* const* d_in, const int* in_sizes, int n_in,
                              void* d_out, int out_size, void* d_ws, size_t ws_size,
                              hipStream_t stream) {
    const float* input  = (const float*)d_in[0];   // [8192,1,28,28]
    const float* rand_u = (const float*)d_in[1];   // [T,8192,784]
    const float* W1     = (const float*)d_in[2];   // [1024,784]
    const float* W2     = (const float*)d_in[3];   // [10,1024]
    float* out = (float*)d_out;

    int T = in_sizes[1] / (B_TOT * K_IN);          // 50

    _Float16* W1h = (_Float16*)d_ws;               // 1024*800 f16
    _Float16* W2h = W1h + H_DIM * K_PAD;           // 16*1024 f16

    int nconv = H_DIM * K_PAD;                     // covers both conversions
    convert_weights<<<(nconv + 255) / 256, 256, 0, stream>>>(W1, W2, W1h, W2h);

    snn_fused<<<B_TOT / BT, 256, LDS_BYTES, stream>>>(input, rand_u, W1h, W2h, out, T);
}
